// MultiHeadAttention_21784074125977
// MI455X (gfx1250) — matrix operand order repaired and checked by reading, not compiled, not run on hardware
//
#include <hip/hip_runtime.h>

// ---------------------------------------------------------------------------
// MI455X (gfx1250, wave32) attention block:
//   qkv = Wqkv @ x ; per-head softmax(Q^T K) V ; out = Wfc @ attn + x ; LN(C)
// bf16 WMMA (v_wmma_f32_16x16x32_bf16, f32 accum), double-buffered LDS with
// global_load_async_to_lds_b128 (ASYNCcnt) for the non-transposed tiles.
// Flash attention uses a 128-q tile / 8 waves and ~87KB LDS (CDNA5 WGP has
// 320KB) to halve K/V re-reads per unit of matrix math.
// ---------------------------------------------------------------------------

typedef __bf16 bf16;
typedef bf16  v16bf __attribute__((ext_vector_type(16)));
typedef float v8f   __attribute__((ext_vector_type(8)));

union Frag    { v16bf v; uint4 u[2]; };
union Pack8   { bf16 h[8];  uint4 u; };
union Pack16  { bf16 h[16]; uint4 u[2]; };

#define WMMA_BF16(A, B, C) \
  __builtin_amdgcn_wmma_f32_16x16x32_bf16(false, (A), false, (B), (short)0, (C), false, false)

// CDNA5 async memory->LDS copy (per-lane 16B), tracked on ASYNCcnt.
// LDS destination is the low 32 bits of the generic pointer (= group-segment
// offset per the flat-aperture rules), global address is a per-lane 64b VGPR pair.
#define ASYNC_LOAD_B128(ldsptr, gptr)                                        \
  asm volatile("global_load_async_to_lds_b128 %0, %1, off"                   \
               :: "v"((unsigned)(unsigned long long)(ldsptr)),               \
                  "v"((const void*)(gptr))                                   \
               : "memory")
#define WAIT_ASYNC0() asm volatile("s_wait_asynccnt 0x0" ::: "memory")

// problem constants
constexpr int kB = 2;
constexpr int kC = 1024;
constexpr int kT = 2048;
constexpr int kH = 16;
constexpr int kD = 64;   // head dim

// ---------------------------------------------------------------------------
// fp32 -> bf16 conversion (vectorized x4)
// ---------------------------------------------------------------------------
__global__ void cvt_bf16(const float* __restrict__ in, bf16* __restrict__ out, int n) {
  int i = (blockIdx.x * blockDim.x + threadIdx.x) * 4;
  if (i >= n) return;
  float4 f = *(const float4*)(in + i);
  union { bf16 h[4]; uint2 u; } p;
  p.h[0] = (bf16)f.x; p.h[1] = (bf16)f.y; p.h[2] = (bf16)f.z; p.h[3] = (bf16)f.w;
  *(uint2*)(out + i) = p.u;
}

// ---------------------------------------------------------------------------
// GEMM: out[b,o,t] = sum_c w[o,c] * src[b,c,t]   (K = kC = 1024)
// out^T(t,o) = src^T(t,c) * w^T(c,o):
//   A = src^T (VGPR-staged LDS transpose), B = w (async copy, no transpose),
//   C-layout lanes give 8 contiguous t per lane -> 16B coalesced stores.
// Block tile 128(t) x 128(o) x 32(c); 8 waves in 2(m) x 4(n).
// Ping-pong double buffer: one barrier per k-step; next-step global loads and
// async-LDS copies are issued right after the barrier and overlap the WMMAs.
// ---------------------------------------------------------------------------
template <bool FC>
__global__ __launch_bounds__(256)
void gemm_bf16(const bf16* __restrict__ src,    // [b] kC x kT
               const bf16* __restrict__ w,      // No x kC
               bf16* __restrict__ outb,         // [b] No x kT (FC=false)
               float* __restrict__ outf,        // [b] No x kT (FC=true)
               const float* __restrict__ resid, // x, f32 (FC=true)
               int No) {
  __shared__ __align__(16) bf16 As[2][128 * 40];   // [t][c], pad 8
  __shared__ __align__(16) bf16 Ws[2][128 * 40];   // [o][c], pad 8

  const int tid  = threadIdx.x;
  const int lane = tid & 31, wave = tid >> 5;
  const int wm = wave >> 2, wn = wave & 3;      // 2 x 4 wave grid
  const int half = lane >> 4, l16 = lane & 15;
  const int b  = blockIdx.z;
  const int t0 = blockIdx.x * 128;
  const int o0 = blockIdx.y * 128;
  const bf16* srcb = src + (size_t)b * kC * kT;

  v8f acc[4][2] = {};

  const int ac = tid >> 3;             // c row for A stage (0..31)
  const int at = (tid & 7) * 16;       // t part
  const int wr = tid >> 1;             // o row for W stage (0..127)
  const int wp = (tid & 1) * 16;       // c part

  // ---- prologue: stage k-step 0 ----
  Pack16 xa;
  {
    const bf16* g = srcb + (size_t)ac * kT + t0 + at;
    xa.u[0] = *(const uint4*)g;
    xa.u[1] = *(const uint4*)(g + 8);
    const bf16* gw = w + (size_t)(o0 + wr) * kC + wp;
    ASYNC_LOAD_B128(&Ws[0][wr * 40 + wp],     gw);
    ASYNC_LOAD_B128(&Ws[0][wr * 40 + wp + 8], gw + 8);
  }

  for (int it = 0; it < kC / 32; ++it) {
    const int p = it & 1;
    // publish A^T tile for this step
#pragma unroll
    for (int j = 0; j < 16; j++) As[p][(at + j) * 40 + ac] = xa.h[j];
    WAIT_ASYNC0();        // this wave's Ws[p] async writes are in LDS
    __syncthreads();      // everyone's As[p]/Ws[p] published

    if (it + 1 < kC / 32) {  // issue next step (overlaps with WMMAs below)
      const int kb = (it + 1) * 32;
      const bf16* g = srcb + (size_t)(kb + ac) * kT + t0 + at;
      xa.u[0] = *(const uint4*)g;
      xa.u[1] = *(const uint4*)(g + 8);
      const bf16* gw = w + (size_t)(o0 + wr) * kC + kb + wp;
      ASYNC_LOAD_B128(&Ws[p ^ 1][wr * 40 + wp],     gw);
      ASYNC_LOAD_B128(&Ws[p ^ 1][wr * 40 + wp + 8], gw + 8);
    }

    Frag a[4], bw[2];
#pragma unroll
    for (int i = 0; i < 4; i++) {
      const int tr = wm * 64 + i * 16 + l16;
      a[i].u[0] = *(const uint4*)&As[p][tr * 40 + half * 8];
      a[i].u[1] = *(const uint4*)&As[p][tr * 40 + 16 + half * 8];
    }
#pragma unroll
    for (int j = 0; j < 2; j++) {
      const int oc = wn * 32 + j * 16 + l16;
      bw[j].u[0] = *(const uint4*)&Ws[p][oc * 40 + half * 8];
      bw[j].u[1] = *(const uint4*)&Ws[p][oc * 40 + 16 + half * 8];
    }
#pragma unroll
    for (int i = 0; i < 4; i++)
#pragma unroll
      for (int j = 0; j < 2; j++)
        acc[i][j] = WMMA_BF16(a[i].v, bw[j].v, acc[i][j]);
  }

#pragma unroll
  for (int i = 0; i < 4; i++) {
    const int tbase = t0 + wm * 64 + i * 16 + half * 8;  // 8 contiguous t per lane
#pragma unroll
    for (int j = 0; j < 2; j++) {
      const int o = o0 + wn * 32 + j * 16 + l16;
      const size_t base = (size_t)b * No * kT + (size_t)o * kT + tbase;
      if constexpr (FC) {
        float4 r0 = *(const float4*)(resid + base);
        float4 r1 = *(const float4*)(resid + base + 4);
        float4 v0 = {acc[i][j][0] + r0.x, acc[i][j][1] + r0.y,
                     acc[i][j][2] + r0.z, acc[i][j][3] + r0.w};
        float4 v1 = {acc[i][j][4] + r1.x, acc[i][j][5] + r1.y,
                     acc[i][j][6] + r1.z, acc[i][j][7] + r1.w};
        *(float4*)(outf + base)     = v0;
        *(float4*)(outf + base + 4) = v1;
      } else {
        Pack8 pk;
#pragma unroll
        for (int r = 0; r < 8; r++) pk.h[r] = (bf16)acc[i][j][r];
        *(uint4*)(outb + base) = pk.u;
      }
    }
  }
}

// ---------------------------------------------------------------------------
// Fused flash attention (no 1/sqrt(d) scaling — faithful to reference).
// Block = 256 threads (8 waves); block tile = 128 q rows, wave = 16 q rows.
// S = Q^T K via WMMA; online softmax with width-16 shfl reductions;
// O += P V^T (V staged with async-to-LDS: its reduction dim k is contiguous).
// ~87KB LDS (within CDNA5's 320KB/WGP). Output Cc[b, h*64+d, q], bf16.
// ---------------------------------------------------------------------------
__global__ __launch_bounds__(256)
void flash_attn(const bf16* __restrict__ qkv, bf16* __restrict__ cc) {
  __shared__ __align__(16) bf16 Qs[128 * 72];    // [q][d]      18.4 KB
  __shared__ __align__(16) bf16 Ks[128 * 72];    // [k][d]      18.4 KB
  __shared__ __align__(16) bf16 Vs[64 * 136];    // [d][k]      17.4 KB
  __shared__ __align__(16) bf16 Ps[128 * 136];   // [wv*16+m][k] 34.8 KB

  const int tid  = threadIdx.x;
  const int lane = tid & 31, wave = tid >> 5;
  const int half = lane >> 4, l16 = lane & 15;
  const int q0 = blockIdx.x * 128, h = blockIdx.y, b = blockIdx.z;
  const size_t bs = (size_t)b * 3 * kC * kT;
  const bf16* Qg = qkv + bs + (size_t)(h * kD) * kT;
  const bf16* Kg = qkv + bs + (size_t)(kC + h * kD) * kT;
  const bf16* Vg = qkv + bs + (size_t)(2 * kC + h * kD) * kT;

  const int sd = tid >> 2;             // stage row d (0..63)
  const int sp = (tid & 3) * 32;       // stage col part

  { // stage Q tile (64d x 128q) -> Qs[q][d]
    const bf16* g = Qg + (size_t)sd * kT + q0 + sp;
#pragma unroll
    for (int c = 0; c < 2; c++) {
      Pack16 p;
      p.u[0] = *(const uint4*)(g + c * 16);
      p.u[1] = *(const uint4*)(g + c * 16 + 8);
#pragma unroll
      for (int j = 0; j < 16; j++) Qs[(sp + c * 16 + j) * 72 + sd] = p.h[j];
    }
  }
  __syncthreads();

  Frag aQ[2];
  {
    const int qr = wave * 16 + l16;
#pragma unroll
    for (int c = 0; c < 2; c++) {
      aQ[c].u[0] = *(const uint4*)&Qs[qr * 72 + c * 32 + half * 8];
      aQ[c].u[1] = *(const uint4*)&Qs[qr * 72 + c * 32 + 16 + half * 8];
    }
  }

  v8f oacc[4] = {};
  const v8f vzero = {};
  float mrow[8], lrow[8];
#pragma unroll
  for (int r = 0; r < 8; r++) { mrow[r] = -1e30f; lrow[r] = 0.f; }

  for (int kt = 0; kt < kT; kt += 128) {
    __syncthreads();   // previous tile's compute done -> Ks/Vs reusable
    {
      // V tile -> Vs[d][k]: straight copy, async memory->LDS (ASYNCcnt)
      const bf16* gv = Vg + (size_t)sd * kT + kt + sp;
#pragma unroll
      for (int c = 0; c < 4; c++)
        ASYNC_LOAD_B128(&Vs[sd * 136 + sp + c * 8], gv + c * 8);
      // K tile -> Ks[k][d]: transpose via VGPR round trip (overlaps async)
      const bf16* g = Kg + (size_t)sd * kT + kt + sp;
#pragma unroll
      for (int c = 0; c < 2; c++) {
        Pack16 p;
        p.u[0] = *(const uint4*)(g + c * 16);
        p.u[1] = *(const uint4*)(g + c * 16 + 8);
#pragma unroll
        for (int j = 0; j < 16; j++) Ks[(sp + c * 16 + j) * 72 + sd] = p.h[j];
      }
    }
    WAIT_ASYNC0();
    __syncthreads();

    // ---- S = Q^T K  (16q x 128k per wave) ----
    v8f s[8];
#pragma unroll
    for (int nf = 0; nf < 8; nf++) {
      s[nf] = vzero;
#pragma unroll
      for (int c = 0; c < 2; c++) {
        Frag bk;
        const int kc = nf * 16 + l16;
        bk.u[0] = *(const uint4*)&Ks[kc * 72 + c * 32 + half * 8];
        bk.u[1] = *(const uint4*)&Ks[kc * 72 + c * 32 + 16 + half * 8];
        s[nf] = WMMA_BF16(aQ[c].v, bk.v, s[nf]);
      }
    }

    // ---- online softmax per q row; emit P (bf16) to LDS ----
#pragma unroll
    for (int r = 0; r < 8; r++) {
      float m = -1e30f;
#pragma unroll
      for (int nf = 0; nf < 8; nf++) m = fmaxf(m, s[nf][r]);
#pragma unroll
      for (int off = 1; off < 16; off <<= 1) m = fmaxf(m, __shfl_xor(m, off, 16));
      const float mn   = fmaxf(mrow[r], m);
      const float corr = __expf(mrow[r] - mn);
      mrow[r] = mn;
      lrow[r] *= corr;
#pragma unroll
      for (int n = 0; n < 4; n++) oacc[n][r] *= corr;
      float ls = 0.f;
#pragma unroll
      for (int nf = 0; nf < 8; nf++) {
        const float p = __expf(s[nf][r] - mn);
        ls += p;
        Ps[(wave * 16 + half * 8 + r) * 136 + nf * 16 + l16] = (bf16)p;
      }
#pragma unroll
      for (int off = 1; off < 16; off <<= 1) ls += __shfl_xor(ls, off, 16);
      lrow[r] += ls;
    }

    // ---- O += P * V^T  (16q x 64d per wave) ----
#pragma unroll
    for (int c = 0; c < 4; c++) {
      Frag ap;
      const int pr = wave * 16 + l16;
      ap.u[0] = *(const uint4*)&Ps[pr * 136 + c * 32 + half * 8];
      ap.u[1] = *(const uint4*)&Ps[pr * 136 + c * 32 + 16 + half * 8];
#pragma unroll
      for (int n = 0; n < 4; n++) {
        Frag bv;
        const int dr = n * 16 + l16;
        bv.u[0] = *(const uint4*)&Vs[dr * 136 + c * 32 + half * 8];
        bv.u[1] = *(const uint4*)&Vs[dr * 136 + c * 32 + 16 + half * 8];
        oacc[n] = WMMA_BF16(ap.v, bv.v, oacc[n]);
      }
    }
  }

  // finalize: O /= l ; store Cc[b, h*64+d, q] with 16B coalesced stores
  float inv[8];
#pragma unroll
  for (int r = 0; r < 8; r++) inv[r] = 1.f / lrow[r];
#pragma unroll
  for (int n = 0; n < 4; n++) {
    Pack8 p;
#pragma unroll
    for (int r = 0; r < 8; r++) p.h[r] = (bf16)(oacc[n][r] * inv[r]);
    const int dcol = h * kD + n * 16 + l16;
    const int qb   = q0 + wave * 16 + half * 8;
    *(uint4*)(cc + (size_t)b * kC * kT + (size_t)dcol * kT + qb) = p.u;
  }
}

// ---------------------------------------------------------------------------
// LayerNorm over channel dim per (b,t). 64 columns per block, 4-way channel
// split per column with LDS reduction; all global accesses coalesced.
// ---------------------------------------------------------------------------
__global__ __launch_bounds__(256)
void layernorm_ch(const float* __restrict__ in,
                  const float* __restrict__ gamma,
                  const float* __restrict__ beta,
                  float* __restrict__ out) {
  __shared__ float red[2][4][64];
  const int tx = threadIdx.x & 63, ty = threadIdx.x >> 6;
  const int col = blockIdx.x * 64 + tx;           // 0 .. B*T-1
  const int b = col >> 11, t = col & (kT - 1);
  const float* p = in + (size_t)b * kC * kT + t;
  float s = 0.f, s2 = 0.f;
  for (int c = ty * 256; c < ty * 256 + 256; c++) {
    const float v = p[(size_t)c * kT];
    s += v; s2 += v * v;
  }
  red[0][ty][tx] = s; red[1][ty][tx] = s2;
  __syncthreads();
  float S = 0.f, S2 = 0.f;
#pragma unroll
  for (int j = 0; j < 4; j++) { S += red[0][j][tx]; S2 += red[1][j][tx]; }
  const float mu   = S * (1.f / kC);
  const float var  = S2 * (1.f / kC) - mu * mu;
  const float rstd = rsqrtf(var + 1e-6f);
  float* q = out + (size_t)b * kC * kT + t;
  for (int c = ty * 256; c < ty * 256 + 256; c++) {
    const float v = p[(size_t)c * kT];
    q[(size_t)c * kT] = (v - mu) * rstd * gamma[c] + beta[c];
  }
}

// ---------------------------------------------------------------------------
// launch
// ---------------------------------------------------------------------------
extern "C" void kernel_launch(void* const* d_in, const int* in_sizes, int n_in,
                              void* d_out, int out_size, void* d_ws, size_t ws_size,
                              hipStream_t stream) {
  const float* x     = (const float*)d_in[0];
  const float* wqkv  = (const float*)d_in[1];
  const float* wfc   = (const float*)d_in[2];
  const float* gamma = (const float*)d_in[3];
  const float* beta  = (const float*)d_in[4];
  float* out = (float*)d_out;

  char* ws = (char*)d_ws;
  // workspace layout (bytes)
  bf16*  xb    = (bf16*)(ws);                          //  8,388,608  x bf16
  bf16*  wqkvb = (bf16*)(ws + 8388608);                //  6,291,456  Wqkv bf16
  bf16*  wfcb  = (bf16*)(ws + 14680064);               //  2,097,152  Wfc bf16
  bf16*  qkvb  = (bf16*)(ws + 16777216);               // 25,165,824  qkv bf16
  bf16*  ccb   = (bf16*)(ws + 41943040);               //  8,388,608  attn out bf16
  float* outf  = (float*)(ws + 50331648);              // 16,777,216  pre-LN f32

  const int nx = kB * kC * kT;        // 4,194,304
  const int n1 = 3 * kC * kC;         // 3,145,728
  const int n2 = kC * kC;             // 1,048,576

  cvt_bf16<<<(nx / 4 + 255) / 256, 256, 0, stream>>>(x, xb, nx);
  cvt_bf16<<<(n1 / 4 + 255) / 256, 256, 0, stream>>>(wqkv, wqkvb, n1);
  cvt_bf16<<<(n2 / 4 + 255) / 256, 256, 0, stream>>>(wfc, wfcb, n2);

  // qkv = Wqkv @ x : grid (T/128, 3C/128, B)
  gemm_bf16<false><<<dim3(kT / 128, 3 * kC / 128, kB), 256, 0, stream>>>(
      xb, wqkvb, qkvb, nullptr, nullptr, 3 * kC);

  // fused attention: grid (T/128 q-tiles, H, B)
  flash_attn<<<dim3(kT / 128, kH, kB), 256, 0, stream>>>(qkvb, ccb);

  // out = Wfc @ cc + x : grid (T/128, C/128, B)
  gemm_bf16<true><<<dim3(kT / 128, kC / 128, kB), 256, 0, stream>>>(
      ccb, wfcb, nullptr, outf, x, kC);

  // LayerNorm over channels
  layernorm_ch<<<(kB * kT) / 64, 256, 0, stream>>>(outf, gamma, beta, out);
}
